// Encoder_64604898066521
// MI455X (gfx1250) — compile-verified, matrix operand order checked by reading, not run on hardware
//
#include <hip/hip_runtime.h>
#include <math.h>

// ---------------- types ----------------
typedef __attribute__((ext_vector_type(16))) __bf16 v16bf;
typedef __attribute__((ext_vector_type(8)))  __bf16 v8bf;
typedef __attribute__((ext_vector_type(2)))  __bf16 v2bf;
typedef __attribute__((ext_vector_type(8)))  float  v8f;
typedef __attribute__((ext_vector_type(2)))  float  v2f;

#define B_SZ   128
#define NTOK   196
#define NVIS   49
#define NMASK  147
#define DM     768
#define NH     12
#define HDIM   64
#define FF     3072
#define NLAYER 12

#define BM 128
#define BN 128
#define BK 32

// ---------------- mask / visible-index computation ----------------
__global__ void k_mask_vis(const int* __restrict__ perm, int* flag, int* vis_idx,
                           int* vis_pos, float* mask_out, int write_mask) {
  if (threadIdx.x == 0 && blockIdx.x == 0) {
    for (int n = 0; n < NTOK; n++) flag[n] = 0;
    for (int i = 0; i < NMASK; i++) flag[perm[NVIS + i]] = 1;
    int j = 0;
    for (int n = 0; n < NTOK; n++) {
      if (!flag[n]) { vis_pos[n] = j; vis_idx[j] = n; j++; }
      else vis_pos[n] = -1;
      if (write_mask) mask_out[n] = flag[n] ? 1.0f : 0.0f;
    }
  }
}

// ---------------- patchify + LayerNorm (wave per token row) ----------------
__global__ __launch_bounds__(256) void k_patchify_ln(
    const float* __restrict__ img, const float* __restrict__ g,
    const float* __restrict__ bta, float* __restrict__ out) {
  int lane = threadIdx.x & 31;
  int wid  = threadIdx.x >> 5;
  int t = blockIdx.x * 8 + wid;          // token row 0..25087
  if (t >= B_SZ * NTOK) return;
  int b = t / NTOK, n = t % NTOK;
  int ph = n / 14, pw = n % 14;
  float v[24];
  float s = 0.f, ss = 0.f;
#pragma unroll
  for (int j = 0; j < 24; j++) {
    int e = lane + j * 32;               // 0..767
    int pi = e / 3, c = e % 3;
    int py = pi >> 4, px = pi & 15;
    float x = img[(((size_t)b * 3 + c) * 224 + ph * 16 + py) * 224 + pw * 16 + px];
    v[j] = x; s += x; ss += x * x;
  }
#pragma unroll
  for (int o = 16; o > 0; o >>= 1) { s += __shfl_xor(s, o); ss += __shfl_xor(ss, o); }
  float mu = s * (1.0f / DM);
  float var = ss * (1.0f / DM) - mu * mu;
  float inv = rsqrtf(var + 1e-5f);
#pragma unroll
  for (int j = 0; j < 24; j++) {
    int e = lane + j * 32;
    out[(size_t)t * DM + e] = (v[j] - mu) * inv * g[e] + bta[e];
  }
}

// ---------------- generic LayerNorm (wave per row of 768) ----------------
__global__ __launch_bounds__(256) void k_ln(
    const float* __restrict__ x, const float* __restrict__ g,
    const float* __restrict__ bta, float* __restrict__ y, int rows) {
  int lane = threadIdx.x & 31;
  int wid  = threadIdx.x >> 5;
  int t = blockIdx.x * 8 + wid;
  if (t >= rows) return;
  const float* xr = x + (size_t)t * DM;
  float v[24];
  float s = 0.f, ss = 0.f;
#pragma unroll
  for (int j = 0; j < 24; j++) {
    float q = xr[lane + j * 32];
    v[j] = q; s += q; ss += q * q;
  }
#pragma unroll
  for (int o = 16; o > 0; o >>= 1) { s += __shfl_xor(s, o); ss += __shfl_xor(ss, o); }
  float mu = s * (1.0f / DM);
  float var = ss * (1.0f / DM) - mu * mu;
  float inv = rsqrtf(var + 1e-5f);
#pragma unroll
  for (int j = 0; j < 24; j++) {
    int e = lane + j * 32;
    y[(size_t)t * DM + e] = (v[j] - mu) * inv * g[e] + bta[e];
  }
}

// ---------------- LN2 + pos-enc + visible gather (wave per visible token) --
__global__ __launch_bounds__(256) void k_ln2_posenc(
    const float* __restrict__ x1, const float* __restrict__ g,
    const float* __restrict__ bta, const int* __restrict__ vis_idx,
    float* __restrict__ h) {
  int lane = threadIdx.x & 31;
  int wid  = threadIdx.x >> 5;
  int t = blockIdx.x * 8 + wid;          // 0..6271
  if (t >= B_SZ * NVIS) return;
  int b = t / NVIS, i = t % NVIS;
  int n = vis_idx[i];
  const float* xr = x1 + ((size_t)b * NTOK + n) * DM;
  float v[24];
  float s = 0.f, ss = 0.f;
#pragma unroll
  for (int j = 0; j < 24; j++) {
    float q = xr[lane + j * 32];
    v[j] = q; s += q; ss += q * q;
  }
#pragma unroll
  for (int o = 16; o > 0; o >>= 1) { s += __shfl_xor(s, o); ss += __shfl_xor(ss, o); }
  float mu = s * (1.0f / DM);
  float var = ss * (1.0f / DM) - mu * mu;
  float inv = rsqrtf(var + 1e-5f);
#pragma unroll
  for (int j = 0; j < 24; j++) {
    int e = lane + j * 32;
    int i2 = e >> 1;
    float angle = (float)n * __expf(-(float)i2 * (9.210340371976184f / 384.0f));
    float pe = (e & 1) ? __cosf(angle) : __sinf(angle);
    h[(size_t)t * DM + e] = (v[j] - mu) * inv * g[e] + bta[e] + pe;
  }
}

// ---------------- bf16 pack helpers (true pair converts) ----------------
__device__ __forceinline__ unsigned pack2_bf16(float x, float y) {
  v2bf r = __builtin_convertvector((v2f){x, y}, v2bf);   // v_cvt_pk_bf16_f32
  union { v2bf b; unsigned u; } c;
  c.b = r;
  return c.u;
}
__device__ __forceinline__ uint2 pack4_bf16(float4 f) {
  uint2 u;
  u.x = pack2_bf16(f.x, f.y);
  u.y = pack2_bf16(f.z, f.w);
  return u;
}

// ---------------- WMMA fragment load helper ----------------
// LDS subtile layout: 16 rows (m or n) x 32 k, row-major bf16.
// lanes 0-15:  row = lane,    elems 0..7 = k0..7,  8..15 = k16..23
// lanes 16-31: row = lane-16, elems 0..7 = k8..15, 8..15 = k24..31
__device__ __forceinline__ v16bf load_frag(const __bf16* base, int lane) {
  int r  = lane & 15;
  int kh = (lane >> 4) * 8;
  const __bf16* p = base + r * BK + kh;
  v8bf lo = *(const v8bf*)(p);
  v8bf hi = *(const v8bf*)(p + 16);
  v16bf f;
#pragma unroll
  for (int i = 0; i < 8; i++) { f[i] = lo[i]; f[8 + i] = hi[i]; }
  return f;
}

// ---------------- bf16 WMMA GEMM: C = act(A@W + bias) + resid -------------
// A: MxK f32 row-major; W: KxN f32 row-major; all dims multiples of tile.
// Double-buffered LDS, register-staged prefetch, one barrier per K-step.
__global__ __launch_bounds__(256) void k_gemm(
    const float* __restrict__ A, const float* __restrict__ W,
    const float* __restrict__ bias, const float* resid,
    float* C, int M, int K, int N, int act) {
  __shared__ __align__(16) __bf16 As[2][BM * BK];
  __shared__ __align__(16) __bf16 Bs[2][BN * BK];

  int tid  = threadIdx.x;
  int lane = tid & 31;
  int wid  = tid >> 5;
  int bm = blockIdx.y * BM;
  int bn = blockIdx.x * BN;
  int wm = (wid >> 2) * 64;      // 0 or 64
  int wn = (wid & 3) * 32;       // 0,32,64,96

  v8f acc[4][2] = {};
  float4 ar[4];
  float4 br0[2], br1[2];

  // A: idx = tid + i*256 over 1024 float4 groups: m = idx>>3, kq = idx&7
  // W: idx = tid + i*256 over 512 (2k-row x 4n) groups: k2 = idx>>5, n4 = (idx&31)*4
  auto gload = [&](int k0) {
#pragma unroll
    for (int i = 0; i < 4; i++) {
      int idx = tid + i * 256;
      int m = idx >> 3, kq = idx & 7;
      ar[i] = *(const float4*)(A + (size_t)(bm + m) * K + k0 + kq * 4);
    }
#pragma unroll
    for (int i = 0; i < 2; i++) {
      int idx = tid + i * 256;
      int k2 = idx >> 5, n4 = (idx & 31) * 4;
      const float* p0 = W + (size_t)(k0 + k2 * 2) * N + bn + n4;
      br0[i] = *(const float4*)(p0);
      br1[i] = *(const float4*)(p0 + N);
    }
  };
  auto sstore = [&](int buf) {
#pragma unroll
    for (int i = 0; i < 4; i++) {
      int idx = tid + i * 256;
      int m = idx >> 3, kq = idx & 7;
      *(uint2*)(&As[buf][m * BK + kq * 4]) = pack4_bf16(ar[i]);
    }
#pragma unroll
    for (int i = 0; i < 2; i++) {
      int idx = tid + i * 256;
      int k2 = idx >> 5, n4 = (idx & 31) * 4;
      int kb = k2 * 2;
      // pairs along k at fixed n -> single packed b32 store into [n][k] layout
      *(unsigned*)(&Bs[buf][(n4 + 0) * BK + kb]) = pack2_bf16(br0[i].x, br1[i].x);
      *(unsigned*)(&Bs[buf][(n4 + 1) * BK + kb]) = pack2_bf16(br0[i].y, br1[i].y);
      *(unsigned*)(&Bs[buf][(n4 + 2) * BK + kb]) = pack2_bf16(br0[i].z, br1[i].z);
      *(unsigned*)(&Bs[buf][(n4 + 3) * BK + kb]) = pack2_bf16(br0[i].w, br1[i].w);
    }
  };

  int nk = K / BK;
  gload(0);
  for (int k = 0; k < nk; k++) {
    int buf = k & 1;
    sstore(buf);
    __syncthreads();
    if (k + 1 < nk) gload((k + 1) * BK);   // overlap next global loads with WMMA

    v16bf afr[4], bfr[2];
#pragma unroll
    for (int mi = 0; mi < 4; mi++) afr[mi] = load_frag(&As[buf][(wm + mi * 16) * BK], lane);
#pragma unroll
    for (int ni = 0; ni < 2; ni++) bfr[ni] = load_frag(&Bs[buf][(wn + ni * 16) * BK], lane);

#pragma unroll
    for (int mi = 0; mi < 4; mi++)
#pragma unroll
      for (int ni = 0; ni < 2; ni++)
        acc[mi][ni] = __builtin_amdgcn_wmma_f32_16x16x32_bf16(
            false, afr[mi], false, bfr[ni], (short)0, acc[mi][ni], false, false);
    // next iteration's sstore targets the other buffer; the barrier above
    // already guaranteed every wave finished computing from it.
  }

  // epilogue: C/D layout — VGPR j: lanes0-15 M=j, lanes16-31 M=8+j; N=lane&15
  int rsel = (lane >> 4) * 8;
  int csel = lane & 15;
#pragma unroll
  for (int mi = 0; mi < 4; mi++)
#pragma unroll
    for (int ni = 0; ni < 2; ni++)
#pragma unroll
      for (int j = 0; j < 8; j++) {
        int row = bm + wm + mi * 16 + rsel + j;
        int col = bn + wn + ni * 16 + csel;
        float v = acc[mi][ni][j];
        if (bias) v += bias[col];
        if (act)  v = 0.5f * v * (1.0f + erff(v * 0.7071067811865475f));
        if (resid) v += resid[(size_t)row * N + col];
        C[(size_t)row * N + col] = v;
      }
}

// ---------------- attention: wave per (b, head, query-row) ----------------
__global__ __launch_bounds__(256) void k_attn(const float* __restrict__ qkv,
                                              float* __restrict__ o) {
  int lane = threadIdx.x & 31;
  int wid  = threadIdx.x >> 5;
  int r = blockIdx.x * 8 + wid;          // 0 .. B*NH*NVIS-1
  if (r >= B_SZ * NH * NVIS) return;
  int qi = r % NVIS;
  int h  = (r / NVIS) % NH;
  int b  = r / (NVIS * NH);
  const float* base = qkv + (size_t)(b * NVIS) * (3 * DM);
  const float* qrow = base + (size_t)qi * (3 * DM) + h * HDIM;

  int m0 = lane, m1 = lane + 32;
  bool has1 = (m1 < NVIS);
  const float* k0p = base + (size_t)m0 * (3 * DM) + DM + h * HDIM;
  const float* k1p = has1 ? base + (size_t)m1 * (3 * DM) + DM + h * HDIM : k0p;
  float s0 = 0.f, s1 = 0.f;
#pragma unroll 8
  for (int d = 0; d < HDIM; d++) {
    float qd = qrow[d];
    s0 += qd * k0p[d];
    if (has1) s1 += qd * k1p[d];
  }
  s0 *= 0.125f; s1 *= 0.125f;            // 1/sqrt(64)
  float e1m = has1 ? s1 : -3.4e38f;
  float mx = fmaxf(s0, e1m);
#pragma unroll
  for (int off = 16; off > 0; off >>= 1) mx = fmaxf(mx, __shfl_xor(mx, off));
  float w0 = __expf(s0 - mx);
  float w1 = has1 ? __expf(s1 - mx) : 0.f;
  float sum = w0 + w1;
#pragma unroll
  for (int off = 16; off > 0; off >>= 1) sum += __shfl_xor(sum, off);
  float inv = 1.0f / sum;
  w0 *= inv; w1 *= inv;

  float o0 = 0.f, o1 = 0.f;
  for (int m = 0; m < NVIS; m++) {
    float w = (m < 32) ? __shfl(w0, m) : __shfl(w1, m - 32);
    const float* vp = base + (size_t)m * (3 * DM) + 2 * DM + h * HDIM;
    o0 += w * vp[lane];
    o1 += w * vp[lane + 32];
  }
  float* orow = o + (size_t)(b * NVIS + qi) * DM + h * HDIM;
  orow[lane] = o0;
  orow[lane + 32] = o1;
}

// ---------------- final scatter: mask_token canvas + visible rows ---------
__global__ __launch_bounds__(256) void k_output(
    const float* __restrict__ h, const float* __restrict__ mask_token,
    const int* __restrict__ vis_pos, float* __restrict__ y) {
  int lane = threadIdx.x & 31;
  int wid  = threadIdx.x >> 5;
  int t = blockIdx.x * 8 + wid;          // 0..25087
  if (t >= B_SZ * NTOK) return;
  int b = t / NTOK, n = t % NTOK;
  int vp = vis_pos[n];
  const float* src = (vp >= 0) ? h + (size_t)(b * NVIS + vp) * DM : mask_token;
#pragma unroll
  for (int j = 0; j < 24; j++) {
    int e = lane + j * 32;
    y[(size_t)t * DM + e] = src[e];
  }
}

// ---------------- host orchestration ----------------
extern "C" void kernel_launch(void* const* d_in, const int* in_sizes, int n_in,
                              void* d_out, int out_size, void* d_ws, size_t ws_size,
                              hipStream_t stream) {
  const float* image    = (const float*)d_in[0];
  const int*   perm     = (const int*)d_in[1];
  const float* pe_ln1_g = (const float*)d_in[2];
  const float* pe_ln1_b = (const float*)d_in[3];
  const float* pe_w     = (const float*)d_in[4];
  const float* pe_b     = (const float*)d_in[5];
  const float* pe_ln2_g = (const float*)d_in[6];
  const float* pe_ln2_b = (const float*)d_in[7];
  const float* ln1_g    = (const float*)d_in[8];
  const float* ln1_b    = (const float*)d_in[9];
  const float* qkv_w    = (const float*)d_in[10];
  const float* out_w    = (const float*)d_in[11];
  const float* ln2_g    = (const float*)d_in[12];
  const float* ln2_b    = (const float*)d_in[13];
  const float* mlp_w1   = (const float*)d_in[14];
  const float* mlp_b1   = (const float*)d_in[15];
  const float* mlp_w2   = (const float*)d_in[16];
  const float* mlp_b2   = (const float*)d_in[17];
  const float* mask_tok = (const float*)d_in[18];

  char* ws = (char*)d_ws;
  size_t off = 0;
  auto alloc = [&](size_t bytes) { size_t o = off; off = (off + bytes + 255) & ~(size_t)255; return o; };
  const size_t BIG = (size_t)25088 * 768 * 4;            // 77,070,336 B
  float* r0   = (float*)(ws + alloc(BIG));               // patch-embed A / MLP hidden
  float* r1   = (float*)(ws + alloc(BIG));               // X1 / qkv
  float* hbuf = (float*)(ws + alloc((size_t)6272 * 768 * 4));
  float* ybuf = (float*)(ws + alloc((size_t)6272 * 768 * 4));
  float* obuf = (float*)(ws + alloc((size_t)6272 * 768 * 4));
  int*   flag    = (int*)(ws + alloc(4096));
  int*   vis_idx = flag + 256;
  int*   vis_pos = flag + 512;

  float* out_y = (float*)d_out;
  const size_t Y_ELEMS = (size_t)B_SZ * NTOK * DM;       // 19,267,584
  int write_mask = ((size_t)out_size >= Y_ELEMS + NTOK) ? 1 : 0;

  k_mask_vis<<<1, 1, 0, stream>>>(perm, flag, vis_idx, vis_pos, out_y + Y_ELEMS, write_mask);

  // patch embed: LN -> GEMM(+bias) -> LN + pos-enc + gather visible
  k_patchify_ln<<<3136, 256, 0, stream>>>(image, pe_ln1_g, pe_ln1_b, r0);
  k_gemm<<<dim3(768 / BN, 25088 / BM), 256, 0, stream>>>(
      r0, pe_w, pe_b, nullptr, r1, 25088, 768, 768, 0);
  k_ln2_posenc<<<784, 256, 0, stream>>>(r1, pe_ln2_g, pe_ln2_b, vis_idx, hbuf);

  for (int l = 0; l < NLAYER; l++) {
    k_ln<<<784, 256, 0, stream>>>(hbuf, ln1_g + l * DM, ln1_b + l * DM, ybuf, 6272);
    k_gemm<<<dim3(2304 / BN, 6272 / BM), 256, 0, stream>>>(
        ybuf, qkv_w + (size_t)l * DM * 3 * DM, nullptr, nullptr, r1, 6272, DM, 3 * DM, 0);
    k_attn<<<9408, 256, 0, stream>>>(r1, obuf);
    k_gemm<<<dim3(768 / BN, 6272 / BM), 256, 0, stream>>>(
        obuf, out_w + (size_t)l * DM * DM, nullptr, hbuf, hbuf, 6272, DM, DM, 0);
    k_ln<<<784, 256, 0, stream>>>(hbuf, ln2_g + l * DM, ln2_b + l * DM, ybuf, 6272);
    k_gemm<<<dim3(FF / BN, 6272 / BM), 256, 0, stream>>>(
        ybuf, mlp_w1 + (size_t)l * DM * FF, mlp_b1 + l * FF, nullptr, r0, 6272, DM, FF, 1);
    k_gemm<<<dim3(768 / BN, 6272 / BM), 256, 0, stream>>>(
        r0, mlp_w2 + (size_t)l * FF * DM, mlp_b2 + l * DM, hbuf, hbuf, 6272, FF, DM, 1);
  }

  k_output<<<3136, 256, 0, stream>>>(hbuf, mask_tok, vis_pos, out_y);
}
